// CLIPWithCMPLoss_31791347925636
// MI455X (gfx1250) — matrix operand order repaired and checked by reading, not compiled, not run on hardware
//
#include <hip/hip_runtime.h>

// ---------------------------------------------------------------------------
// CLIP + CMP loss, MI455X (gfx1250), wave32.
//  - bf16 WMMA 16x16x32 (f32 accum) for all GEMMs
//  - Tensor Data Mover (tensor_load_to_lds) stages each workgroup's 16-row
//    A-stripe in LDS once (TENSORcnt-tracked), instead of re-fetching it from
//    L2 for every j-tile; A-fragments then come from ds_load.
//  - 64 MB logits matrix never materialized: two WMMA streaming passes
//    (rowmax, then fused sumexp + CMP stats). Workspace ~22.6 MB.
// ---------------------------------------------------------------------------

typedef __bf16          v16bf __attribute__((ext_vector_type(16)));
typedef float           v8f   __attribute__((ext_vector_type(8)));
typedef unsigned short  v16us __attribute__((ext_vector_type(16)));
typedef unsigned int    v4u   __attribute__((ext_vector_type(4)));
typedef int             v8i   __attribute__((ext_vector_type(8)));
typedef int             v4i   __attribute__((ext_vector_type(4)));

#define BATCH 4096
#define DIN   768
#define EDIM  512

__device__ __forceinline__ unsigned short f32_to_bf16(float f) {
    unsigned u = __float_as_uint(f);
    u += 0x7FFFu + ((u >> 16) & 1u);          // round-to-nearest-even
    return (unsigned short)(u >> 16);
}
__device__ __forceinline__ float bf16_to_f32(unsigned short h) {
    return __uint_as_float(((unsigned)h) << 16);
}

// ---------------------------------------------------------------------------
// TDM: async 2-D tile load Global -> LDS. D# built per CDNA5 ISA ch.8:
//   group0: [1:0]count=1 | [63:32]lds_addr | [120:64]global_addr | [127:126]type=2
//   group1: [17:16]data_size | [79:48]tensor_dim0 | [111:80]tensor_dim1
//           | [127:112]tile_dim0 | [143:128]tile_dim1 | [207:160]dim0_stride
//   groups 2/3: zero (<=2D tensor)
// dim0 = contiguous elements per row, dim1 = rows, stride in elements.
__device__ __forceinline__ void tdm_load_2d_to_lds(unsigned lds_off, const void* gptr,
                                                   unsigned data_size_code, // 1 => 2 bytes
                                                   unsigned dim0, unsigned dim1,
                                                   unsigned stride0) {
    unsigned long long ga = (unsigned long long)(uintptr_t)gptr;
    v4u g0;
    g0.x = 0x1u;                                            // count=1, user D#
    g0.y = lds_off;                                         // lds_addr (bytes)
    g0.z = (unsigned)ga;                                    // global_addr[31:0]
    g0.w = (unsigned)((ga >> 32) & 0x1FFFFFFu) | 0x80000000u; // addr[56:32] | type=2
    v8i g1;
    g1[0] = (int)(data_size_code << 16);                    // wg_mask=0 (not in cluster)
    g1[1] = (int)((dim0 & 0xFFFFu) << 16);                  // tensor_dim0[15:0]
    g1[2] = (int)(((dim0 >> 16) & 0xFFFFu) | ((dim1 & 0xFFFFu) << 16));
    g1[3] = (int)(((dim1 >> 16) & 0xFFFFu) | ((dim0 & 0xFFFFu) << 16)); // tile_dim0
    g1[4] = (int)(dim1 & 0xFFFFu);                          // tile_dim1 (tile_dim2=0)
    g1[5] = (int)stride0;                                   // tensor_dim0_stride[31:0]
    g1[6] = 0;
    g1[7] = 0;
    v4i gz = {0, 0, 0, 0};
#if __has_include(<hip/amd_detail/amd_gfx1250_TDM.h>)
    v8i gz8 = {0, 0, 0, 0, 0, 0, 0, 0};                     // therock: 6-arg builtin
    __builtin_amdgcn_tensor_load_to_lds(g0, g1, gz, gz, gz8, 0);
#else
    __builtin_amdgcn_tensor_load_to_lds(g0, g1, gz, gz, 0); // ROCm 7.2: 5-arg builtin
#endif
}

__device__ __forceinline__ unsigned lds_offset_of(const void* shared_ptr) {
    // LDS aperture: low 32 bits of a generic pointer to __shared__ = LDS offset
    return (unsigned)(uintptr_t)shared_ptr;
}

// ---------------------------------------------------------------------------
// A fragment: 16x32 bf16 from LDS-staged row-major stripe (16 rows x ld).
// lane L: row = L&15; K = k0 + 0..15 (lo half-wave) or k0+16..31 (hi).
__device__ __forceinline__ v16bf lds_a_frag(const unsigned short* As, int ld, int k0, int lane) {
    const unsigned short* p = As + (lane & 15) * ld + k0 + ((lane >> 4) << 4);
    v16us t;
#pragma unroll
    for (int i = 0; i < 8; ++i) { t[i] = p[i]; t[i + 8] = p[8 + i]; }
    return __builtin_bit_cast(v16bf, t);
}

// B fragment: 32x16 bf16 (KxN); source is N-major ("BT": row n holds K-contig data).
// lane L: col = n0 + (L&15); K = k0 + 0..15 (lo half-wave) or k0+16..31 (hi).
__device__ __forceinline__ v16bf load_b_frag(const unsigned short* __restrict__ BT,
                                             int ld, int n0, int k0, int lane) {
    const unsigned short* p = BT + (size_t)(n0 + (lane & 15)) * ld + k0 + ((lane >> 4) << 4);
    v16us t;
#pragma unroll
    for (int i = 0; i < 16; ++i) t[i] = p[i];
    return __builtin_bit_cast(v16bf, t);
}

// ---------------------------------------------------------------------------
__global__ void __launch_bounds__(256) k_cvt_bf16(const float* __restrict__ src,
                                                  unsigned short* __restrict__ dst, int n) {
    for (int i = blockIdx.x * blockDim.x + threadIdx.x; i < n; i += gridDim.x * blockDim.x)
        dst[i] = f32_to_bf16(src[i]);
}

// W [K x N] f32 -> WT [N x K] bf16 (so B-fragments load contiguously)
__global__ void __launch_bounds__(256) k_cvt_transpose_bf16(const float* __restrict__ W,
                                                            unsigned short* __restrict__ WT,
                                                            int K, int N) {
    int total = K * N;
    for (int i = blockIdx.x * blockDim.x + threadIdx.x; i < total; i += gridDim.x * blockDim.x) {
        int k = i / N, n = i - k * N;
        WT[(size_t)n * K + k] = f32_to_bf16(W[i]);
    }
}

// ---------------------------------------------------------------------------
// P = X[4096x768] @ W[768x512], then L2-normalize rows, emit bf16.
// One workgroup = 16 rows x all 512 cols. 8 waves x (4 tiles of 16 cols).
// X stripe (16x768 bf16 = 24 KB) staged in LDS by TDM once.
__global__ void __launch_bounds__(256) k_proj_norm(const unsigned short* __restrict__ X,
                                                   const unsigned short* __restrict__ WT,
                                                   unsigned short* __restrict__ Out) {
    __shared__ unsigned short Astage[16 * DIN];
    __shared__ float ss[16];
    const int tid = threadIdx.x, wave = tid >> 5, lane = tid & 31;
    const int m0 = blockIdx.x * 16;
    const int nbase = wave * 64;

    if (wave == 0) {
        tdm_load_2d_to_lds(lds_offset_of(Astage), X + (size_t)m0 * DIN,
                           /*2B*/ 1, DIN, 16, DIN);
    }
    if (tid < 16) ss[tid] = 0.0f;
    if (wave == 0) __builtin_amdgcn_s_wait_tensorcnt(0);
    __syncthreads();

    v8f acc[4];
#pragma unroll
    for (int t = 0; t < 4; ++t)
#pragma unroll
        for (int r = 0; r < 8; ++r) acc[t][r] = 0.0f;

    for (int k0 = 0; k0 < DIN; k0 += 32) {
        if (k0 + 32 < DIN)
            __builtin_prefetch(WT + (size_t)(nbase + (lane & 15)) * DIN + k0 + 32, 0, 0);
        v16bf a = lds_a_frag(Astage, DIN, k0, lane);
#pragma unroll
        for (int t = 0; t < 4; ++t) {
            v16bf b = load_b_frag(WT, DIN, nbase + 16 * t, k0, lane);
            acc[t] = __builtin_amdgcn_wmma_f32_16x16x32_bf16(
                false, a, false, b, (short)0, acc[t], false, false);
        }
    }

    const int rh = (lane >> 4) * 8;   // row-half offset from C layout
#pragma unroll
    for (int t = 0; t < 4; ++t)
#pragma unroll
        for (int r = 0; r < 8; ++r) {
            float v = acc[t][r];
            atomicAdd(&ss[r + rh], v * v);
        }
    __syncthreads();

#pragma unroll
    for (int t = 0; t < 4; ++t)
#pragma unroll
        for (int r = 0; r < 8; ++r) {
            int row = r + rh;
            float inv = rsqrtf(ss[row]);
            Out[(size_t)(m0 + row) * EDIM + nbase + 16 * t + (lane & 15)] =
                f32_to_bf16(acc[t][r] * inv);
        }
}

// ---------------------------------------------------------------------------
// Pass 1 over logits: rowmax_i = max_j scale * <imgN_i, txtN_j>
// img stripe (16x512 bf16 = 16 KB) staged in LDS by TDM; reused for all 64 j-tiles.
__global__ void __launch_bounds__(256) k_rowmax(const unsigned short* __restrict__ imgN,
                                                const unsigned short* __restrict__ txtN,
                                                const float* __restrict__ logit_scale,
                                                float* __restrict__ rowmax) {
    __shared__ unsigned short Astage[16 * EDIM];
    __shared__ float part[8][16];
    const int tid = threadIdx.x, wave = tid >> 5, lane = tid & 31;
    const int m0 = blockIdx.x * 16;
    const float scale = __expf(logit_scale[0]);
    const int rh = (lane >> 4) * 8;

    if (wave == 0) {
        tdm_load_2d_to_lds(lds_offset_of(Astage), imgN + (size_t)m0 * EDIM,
                           /*2B*/ 1, EDIM, 16, EDIM);
        __builtin_amdgcn_s_wait_tensorcnt(0);
    }
    __syncthreads();

    float rmax[8];
#pragma unroll
    for (int r = 0; r < 8; ++r) rmax[r] = -3.4e38f;

    for (int jt = wave; jt < BATCH / 64; jt += 8) {
        const int nb = jt * 64;
        v8f acc[4];
#pragma unroll
        for (int t = 0; t < 4; ++t)
#pragma unroll
            for (int r = 0; r < 8; ++r) acc[t][r] = 0.0f;

        for (int k0 = 0; k0 < EDIM; k0 += 32) {
            v16bf a = lds_a_frag(Astage, EDIM, k0, lane);
#pragma unroll
            for (int t = 0; t < 4; ++t) {
                v16bf b = load_b_frag(txtN, EDIM, nb + 16 * t, k0, lane);
                acc[t] = __builtin_amdgcn_wmma_f32_16x16x32_bf16(
                    false, a, false, b, (short)0, acc[t], false, false);
            }
        }
#pragma unroll
        for (int t = 0; t < 4; ++t)
#pragma unroll
            for (int r = 0; r < 8; ++r) rmax[r] = fmaxf(rmax[r], scale * acc[t][r]);
    }

    // reduce across the 16-lane column group
#pragma unroll
    for (int r = 0; r < 8; ++r)
#pragma unroll
        for (int m = 1; m < 16; m <<= 1)
            rmax[r] = fmaxf(rmax[r], __shfl_xor(rmax[r], m));

    if ((lane & 15) == 0) {
#pragma unroll
        for (int r = 0; r < 8; ++r) part[wave][r + rh] = rmax[r];
    }
    __syncthreads();
    if (tid < 16) {
        float m = part[0][tid];
#pragma unroll
        for (int w = 1; w < 8; ++w) m = fmaxf(m, part[w][tid]);
        rowmax[m0 + tid] = m;
    }
}

// ---------------------------------------------------------------------------
// s_iy[i] = scale * <imgN_i, txtN_{labels[i]}>   (one wave per row)
__global__ void __launch_bounds__(256) k_siy(const unsigned short* __restrict__ imgN,
                                             const unsigned short* __restrict__ txtN,
                                             const int* __restrict__ labels,
                                             const float* __restrict__ logit_scale,
                                             float* __restrict__ siy) {
    const int row = (blockIdx.x * blockDim.x + threadIdx.x) >> 5;
    const int lane = threadIdx.x & 31;
    if (row >= BATCH) return;
    const int y = labels[row];
    const unsigned short* a = imgN + (size_t)row * EDIM;
    const unsigned short* b = txtN + (size_t)y * EDIM;
    float s = 0.0f;
    for (int k = lane; k < EDIM; k += 32)
        s += bf16_to_f32(a[k]) * bf16_to_f32(b[k]);
#pragma unroll
    for (int m = 1; m < 32; m <<= 1) s += __shfl_xor(s, m);
    if (lane == 0) siy[row] = __expf(logit_scale[0]) * s;
}

// ---------------------------------------------------------------------------
// Pass 2 over logits (fused): rowsum_i = sum_j exp(s_ij - rowmax_i),
// rowden_i = sum over {labels[j]!=labels[i] && s_ij > s_iy} of exp(s_ij - rowmax_i),
// rowany_i = any such j.  (p_ij > p_yi  <=>  s_ij > s_iy, softmax is monotone.)
__global__ void __launch_bounds__(256) k_rowsum_cmp(const unsigned short* __restrict__ imgN,
                                                    const unsigned short* __restrict__ txtN,
                                                    const int* __restrict__ labels,
                                                    const float* __restrict__ logit_scale,
                                                    const float* __restrict__ rowmax,
                                                    const float* __restrict__ siy,
                                                    float* __restrict__ rowsum,
                                                    float* __restrict__ rowden,
                                                    float* __restrict__ rowany) {
    __shared__ unsigned short Astage[16 * EDIM];
    __shared__ float psum[8][16], pden[8][16], pany[8][16];
    __shared__ float lrm[16], lsy[16];
    __shared__ int   llb[16];
    const int tid = threadIdx.x, wave = tid >> 5, lane = tid & 31;
    const int m0 = blockIdx.x * 16;

    if (wave == 0) {
        tdm_load_2d_to_lds(lds_offset_of(Astage), imgN + (size_t)m0 * EDIM,
                           /*2B*/ 1, EDIM, 16, EDIM);
    }
    if (tid < 16) {
        lrm[tid] = rowmax[m0 + tid];
        lsy[tid] = siy[m0 + tid];
        llb[tid] = labels[m0 + tid];
    }
    if (wave == 0) __builtin_amdgcn_s_wait_tensorcnt(0);
    __syncthreads();

    const float scale = __expf(logit_scale[0]);
    const int rh = (lane >> 4) * 8;
    float rm[8], sy[8];
    int   lb[8];
#pragma unroll
    for (int r = 0; r < 8; ++r) { rm[r] = lrm[r + rh]; sy[r] = lsy[r + rh]; lb[r] = llb[r + rh]; }

    float rsum[8], rden[8], rany[8];
#pragma unroll
    for (int r = 0; r < 8; ++r) { rsum[r] = 0.0f; rden[r] = 0.0f; rany[r] = 0.0f; }

    for (int jt = wave; jt < BATCH / 64; jt += 8) {
        const int nb = jt * 64;
        v8f acc[4];
#pragma unroll
        for (int t = 0; t < 4; ++t)
#pragma unroll
            for (int r = 0; r < 8; ++r) acc[t][r] = 0.0f;

        for (int k0 = 0; k0 < EDIM; k0 += 32) {
            v16bf a = lds_a_frag(Astage, EDIM, k0, lane);
#pragma unroll
            for (int t = 0; t < 4; ++t) {
                v16bf b = load_b_frag(txtN, EDIM, nb + 16 * t, k0, lane);
                acc[t] = __builtin_amdgcn_wmma_f32_16x16x32_bf16(
                    false, a, false, b, (short)0, acc[t], false, false);
            }
        }
#pragma unroll
        for (int t = 0; t < 4; ++t) {
            const int j = nb + 16 * t + (lane & 15);
            const int labj = labels[j];
#pragma unroll
            for (int r = 0; r < 8; ++r) {
                float s = scale * acc[t][r];
                float e = __expf(s - rm[r]);
                rsum[r] += e;
                if (labj != lb[r] && s > sy[r]) { rden[r] += e; rany[r] = 1.0f; }
            }
        }
    }

#pragma unroll
    for (int r = 0; r < 8; ++r)
#pragma unroll
        for (int m = 1; m < 16; m <<= 1) {
            rsum[r] += __shfl_xor(rsum[r], m);
            rden[r] += __shfl_xor(rden[r], m);
            rany[r] = fmaxf(rany[r], __shfl_xor(rany[r], m));
        }

    if ((lane & 15) == 0) {
#pragma unroll
        for (int r = 0; r < 8; ++r) {
            psum[wave][r + rh] = rsum[r];
            pden[wave][r + rh] = rden[r];
            pany[wave][r + rh] = rany[r];
        }
    }
    __syncthreads();
    if (tid < 16) {
        float s = 0.0f, d = 0.0f, a = 0.0f;
#pragma unroll
        for (int w = 0; w < 8; ++w) { s += psum[w][tid]; d += pden[w][tid]; a = fmaxf(a, pany[w][tid]); }
        rowsum[m0 + tid] = s;
        rowden[m0 + tid] = d;
        rowany[m0 + tid] = a;
    }
}

// ---------------------------------------------------------------------------
__global__ void __launch_bounds__(256) k_final(const float* __restrict__ rowmax,
                                               const float* __restrict__ rowsum,
                                               const float* __restrict__ rowden,
                                               const float* __restrict__ rowany,
                                               const float* __restrict__ siy,
                                               float* __restrict__ out) {
    __shared__ float sclip[256], scmp[256];
    float cl = 0.0f, cm = 0.0f;
    for (int i = threadIdx.x; i < BATCH; i += 256) {
        float lp = siy[i] - rowmax[i] - logf(rowsum[i]);     // log_softmax at label col
        cl += -lp;
        float pyi = __expf(siy[i] - rowmax[i]) / rowsum[i];
        float dp  = rowden[i] / rowsum[i];
        cm += (rowany[i] > 0.5f) ? (pyi / (dp + 1e-10f)) : 0.0f;
    }
    sclip[threadIdx.x] = cl;
    scmp[threadIdx.x]  = cm;
    __syncthreads();
    for (int s = 128; s > 0; s >>= 1) {
        if ((int)threadIdx.x < s) {
            sclip[threadIdx.x] += sclip[threadIdx.x + s];
            scmp[threadIdx.x]  += scmp[threadIdx.x + s];
        }
        __syncthreads();
    }
    if (threadIdx.x == 0)
        out[0] = sclip[0] / (float)BATCH + scmp[0] / (float)BATCH;  // LAMBDA_CMP = 1
}

// ---------------------------------------------------------------------------
extern "C" void kernel_launch(void* const* d_in, const int* in_sizes, int n_in,
                              void* d_out, int out_size, void* d_ws, size_t ws_size,
                              hipStream_t stream) {
    const float* images = (const float*)d_in[0];   // [4096 x 768]
    const float* texts  = (const float*)d_in[1];   // [4096 x 768]
    const int*   labels = (const int*)d_in[2];     // [4096]
    const float* W_img  = (const float*)d_in[3];   // [768 x 512]
    const float* W_txt  = (const float*)d_in[4];   // [768 x 512]
    const float* lscale = (const float*)d_in[5];   // scalar
    float* out = (float*)d_out;

    // --- workspace layout (~22.6 MB) ---
    char* p = (char*)d_ws;
    auto take = [&](size_t bytes) { char* r = p; p += (bytes + 255) & ~(size_t)255; return r; };
    unsigned short* imagesB = (unsigned short*)take((size_t)BATCH * DIN * 2);
    unsigned short* textsB  = (unsigned short*)take((size_t)BATCH * DIN * 2);
    unsigned short* WimgT   = (unsigned short*)take((size_t)EDIM * DIN * 2);
    unsigned short* WtxtT   = (unsigned short*)take((size_t)EDIM * DIN * 2);
    unsigned short* imgN    = (unsigned short*)take((size_t)BATCH * EDIM * 2);
    unsigned short* txtN    = (unsigned short*)take((size_t)BATCH * EDIM * 2);
    float* rowmax = (float*)take((size_t)BATCH * 4);
    float* rowsum = (float*)take((size_t)BATCH * 4);
    float* rowden = (float*)take((size_t)BATCH * 4);
    float* rowany = (float*)take((size_t)BATCH * 4);
    float* siy    = (float*)take((size_t)BATCH * 4);
    (void)ws_size; (void)in_sizes; (void)n_in; (void)out_size;

    // 1) precision conversion
    k_cvt_bf16<<<2048, 256, 0, stream>>>(images, imagesB, BATCH * DIN);
    k_cvt_bf16<<<2048, 256, 0, stream>>>(texts,  textsB,  BATCH * DIN);
    k_cvt_transpose_bf16<<<768, 256, 0, stream>>>(W_img, WimgT, DIN, EDIM);
    k_cvt_transpose_bf16<<<768, 256, 0, stream>>>(W_txt, WtxtT, DIN, EDIM);

    // 2) projections + L2 normalize (WMMA, TDM-staged A)
    k_proj_norm<<<BATCH / 16, 256, 0, stream>>>(imagesB, WimgT, imgN);
    k_proj_norm<<<BATCH / 16, 256, 0, stream>>>(textsB,  WtxtT, txtN);

    // 3) softmax rowmax (WMMA streaming, logits never materialized)
    k_rowmax<<<BATCH / 16, 256, 0, stream>>>(imgN, txtN, lscale, rowmax);

    // 4) label-column logits
    k_siy<<<BATCH / 8, 256, 0, stream>>>(imgN, txtN, labels, lscale, siy);

    // 5) fused sumexp + CMP stats (WMMA streaming, TDM-staged A)
    k_rowsum_cmp<<<BATCH / 16, 256, 0, stream>>>(imgN, txtN, labels, lscale,
                                                 rowmax, siy, rowsum, rowden, rowany);

    // 6) reduce to scalar loss
    k_final<<<1, 256, 0, stream>>>(rowmax, rowsum, rowden, rowany, siy, out);
}
